// LayoutLMWithVisualFeatures_41360535060990
// MI455X (gfx1250) — matrix operand order repaired and checked by reading, not compiled, not run on hardware
//
#include <hip/hip_runtime.h>

// Problem dims (fixed by the reference)
#define B_   64
#define S_   512
#define H_   768
#define R_   256
#define VD_  256
#define NC_  9
#define HID_ 512
#define NCP_ 16   // W_h2 padded to 16 rows of Bt so WMMA runs with EXEC all-1s

typedef __attribute__((ext_vector_type(16))) __bf16 v16bf;
typedef __attribute__((ext_vector_type(8)))  __bf16 v8bf;
typedef __attribute__((ext_vector_type(8)))  float  v8f;

#define WMMA_BF16(a, b, c) \
  __builtin_amdgcn_wmma_f32_16x16x32_bf16(false, (a), false, (b), (short)0, (c), false, false)

// ---------------------------------------------------------------------------
// Fragment loaders for v_wmma_f32_16x16x32_bf16 (contiguous k order, same in A and B)
// A (16x32 row-major, M=lane%16): 16 contiguous K values per lane
//   lanes 0-15 : K = kb+0..15
//   lanes 16-31: K = kb+16..31
// B held as Bt [N,K] (column-major of B): lane n gets 16 CONTIGUOUS K values
//   lanes 0-15 : K = kb+0..15 ; lanes 16-31: K = kb+16..31  (one 32B load)
// C/D: c[j] = C[j + 8*(lane>=16)][lane%15? no: lane&15]
// ---------------------------------------------------------------------------
__device__ __forceinline__ v16bf load_A_frag(const __bf16* __restrict__ A, int lda,
                                             int row, int kbase, int lane) {
  int m    = lane & 15;
  int koff = (lane & 16) ? 16 : 0;
  const __bf16* p = A + (size_t)(row + m) * lda + kbase + koff;
  return *reinterpret_cast<const v16bf*>(p);
}

__device__ __forceinline__ v16bf load_Bt_frag(const __bf16* __restrict__ Bt, int ldb,
                                              int kbase, int col, int lane) {
  int n    = lane & 15;
  int koff = (lane & 16) ? 16 : 0;
  // ldb (=K) and kbase+koff are multiples of 16 -> 32B aligned single vector load
  return *reinterpret_cast<const v16bf*>(Bt + (size_t)(col + n) * ldb + kbase + koff);
}

// ---------------------------------------------------------------------------
// f32 -> bf16 conversion (grid-stride)
// ---------------------------------------------------------------------------
__global__ void cvt_f32_bf16(const float* __restrict__ src, __bf16* __restrict__ dst,
                             long n) {
  long i = (long)blockIdx.x * blockDim.x + threadIdx.x;
  long st = (long)gridDim.x * blockDim.x;
  for (; i < n; i += st) dst[i] = (__bf16)src[i];
}

// Tiled transpose + convert: src f32 [K,N] (+batch) -> dst bf16 [N,K] (+batch)
// block = 256 threads (32x8), 32x32 tiles, LDS padded to kill bank conflicts
__global__ void transpose_cvt(const float* __restrict__ src, __bf16* __restrict__ dst,
                              int K, int N, long srcBatch, long dstBatch) {
  __shared__ __bf16 tile[32][33];
  long sb = (long)blockIdx.z * srcBatch;
  long db = (long)blockIdx.z * dstBatch;
  int nb = blockIdx.x * 32, kb = blockIdx.y * 32;
  int tx = threadIdx.x & 31, ty = threadIdx.x >> 5;
#pragma unroll
  for (int i = ty; i < 32; i += 8)
    tile[i][tx] = (__bf16)src[sb + (size_t)(kb + i) * N + nb + tx];
  __syncthreads();
#pragma unroll
  for (int i = ty; i < 32; i += 8)
    dst[db + (size_t)(nb + i) * K + kb + tx] = tile[tx][i];
}

// W_h2 [512,9] f32 -> Wh2t bf16 [16,512] (rows 9..15 zero)
__global__ void cvt_pad_wh2_t(const float* __restrict__ src, __bf16* __restrict__ dst) {
  int i = blockIdx.x * blockDim.x + threadIdx.x;
  if (i >= NCP_ * HID_) return;
  int n = i / HID_, k = i - n * HID_;
  dst[i] = (__bf16)((n < NC_) ? src[k * NC_ + n] : 0.0f);
}

// ---------------------------------------------------------------------------
// Containment mask (bf16 0/1, exact) + 1/max(cnt,1) per (b,r)
// ---------------------------------------------------------------------------
__global__ void mask_kernel(const int* __restrict__ bbox, const float* __restrict__ roi,
                            const int* __restrict__ amask,
                            __bf16* __restrict__ maskout, float* __restrict__ invcnt) {
  int br = blockIdx.x;
  int b = br / R_, r = br - b * R_;
  int s = threadIdx.x;
  const int*   tb = bbox + ((size_t)b * S_ + s) * 4;
  const float* rb = roi  + ((size_t)b * R_ + r) * 4;
  float x1 = (float)tb[0], y1 = (float)tb[1], x2 = (float)tb[2], y2 = (float)tb[3];
  bool inside = (x1 >= rb[0]) && (x2 <= rb[2]) && (y1 >= rb[1]) && (y2 <= rb[3]);
  bool valid  = (amask[(size_t)b * S_ + s] == 1) && (s >= 1);
  int in = (inside && valid) ? 1 : 0;
  maskout[(size_t)br * S_ + s] = (__bf16)(float)in;

  __shared__ int red[S_];
  red[s] = in;
  __syncthreads();
  for (int off = S_ / 2; off > 0; off >>= 1) {
    if (s < off) red[s] += red[s + off];
    __syncthreads();
  }
  if (s == 0) invcnt[br] = 1.0f / fmaxf((float)red[0], 1.0f);
}

// ---------------------------------------------------------------------------
// GEMM pattern: block = 128 threads (4 waves). Wave computes a 64x16 C panel:
// 4 M-tiles sharing one B fragment per K-step (B load amortized 4x).
// ---------------------------------------------------------------------------

// text[b] = (mask[b] @ seq[b]) * invcnt  -> bf16 [B,R,H]
// grid (H/64, R/64, B)
__global__ void pool_gemm(const __bf16* __restrict__ maskm, const __bf16* __restrict__ seqt,
                          const float* __restrict__ invcnt, __bf16* __restrict__ text) {
  int b    = blockIdx.z;
  int wave = threadIdx.x >> 5, lane = threadIdx.x & 31;
  int colt = blockIdx.x * 64 + wave * 16;
  int rowt = blockIdx.y * 64;
  const __bf16* A  = maskm + (size_t)b * R_ * S_;
  const __bf16* Bt = seqt  + (size_t)b * H_ * S_;
  v8f c0 = {}, c1 = {}, c2 = {}, c3 = {};
  for (int k = 0; k < S_; k += 32) {
    v16bf bf = load_Bt_frag(Bt, S_, k, colt, lane);
    c0 = WMMA_BF16(load_A_frag(A, S_, rowt +  0, k, lane), bf, c0);
    c1 = WMMA_BF16(load_A_frag(A, S_, rowt + 16, k, lane), bf, c1);
    c2 = WMMA_BF16(load_A_frag(A, S_, rowt + 32, k, lane), bf, c2);
    c3 = WMMA_BF16(load_A_frag(A, S_, rowt + 48, k, lane), bf, c3);
  }
  int n = lane & 15, mb = (lane >> 4) * 8;
  v8f acc[4] = {c0, c1, c2, c3};
#pragma unroll
  for (int t = 0; t < 4; ++t)
#pragma unroll
    for (int j = 0; j < 8; ++j) {
      int m = rowt + t * 16 + mb + j;
      float sc = invcnt[(size_t)b * R_ + m];
      text[((size_t)b * R_ + m) * H_ + colt + n] = (__bf16)(acc[t][j] * sc);
    }
}

// vis = visfeats @ W_proj + b_proj  [BR,256]x[256,768] -> bf16
// grid (H/64, BR/64)
__global__ void vis_gemm(const __bf16* __restrict__ A, const __bf16* __restrict__ Bt,
                         const float* __restrict__ bias, __bf16* __restrict__ outp) {
  int wave = threadIdx.x >> 5, lane = threadIdx.x & 31;
  int colt = blockIdx.x * 64 + wave * 16;
  int rowt = blockIdx.y * 64;
  v8f c0 = {}, c1 = {}, c2 = {}, c3 = {};
  for (int k = 0; k < VD_; k += 32) {
    v16bf bf = load_Bt_frag(Bt, VD_, k, colt, lane);
    c0 = WMMA_BF16(load_A_frag(A, VD_, rowt +  0, k, lane), bf, c0);
    c1 = WMMA_BF16(load_A_frag(A, VD_, rowt + 16, k, lane), bf, c1);
    c2 = WMMA_BF16(load_A_frag(A, VD_, rowt + 32, k, lane), bf, c2);
    c3 = WMMA_BF16(load_A_frag(A, VD_, rowt + 48, k, lane), bf, c3);
  }
  int n = lane & 15, mb = (lane >> 4) * 8;
  float bv = bias[colt + n];
  v8f acc[4] = {c0, c1, c2, c3};
#pragma unroll
  for (int t = 0; t < 4; ++t)
#pragma unroll
    for (int j = 0; j < 8; ++j)
      outp[((size_t)(rowt + t * 16 + mb + j)) * H_ + colt + n] = (__bf16)(acc[t][j] + bv);
}

// fused = relu(concat(text,vis) @ W_fuse + b_fuse)  K=1536 -> bf16 [BR,H]
// K-tiles of 32 never straddle the 768 boundary. grid (H/64, BR/64)
__global__ void fuse_gemm(const __bf16* __restrict__ text, const __bf16* __restrict__ vis,
                          const __bf16* __restrict__ Bt, const float* __restrict__ bias,
                          __bf16* __restrict__ outp) {
  int wave = threadIdx.x >> 5, lane = threadIdx.x & 31;
  int colt = blockIdx.x * 64 + wave * 16;
  int rowt = blockIdx.y * 64;
  v8f c0 = {}, c1 = {}, c2 = {}, c3 = {};
  for (int k = 0; k < 2 * H_; k += 32) {
    const __bf16* Asrc = (k < H_) ? text : vis;
    int kk = (k < H_) ? k : (k - H_);
    v16bf bf = load_Bt_frag(Bt, 2 * H_, k, colt, lane);
    c0 = WMMA_BF16(load_A_frag(Asrc, H_, rowt +  0, kk, lane), bf, c0);
    c1 = WMMA_BF16(load_A_frag(Asrc, H_, rowt + 16, kk, lane), bf, c1);
    c2 = WMMA_BF16(load_A_frag(Asrc, H_, rowt + 32, kk, lane), bf, c2);
    c3 = WMMA_BF16(load_A_frag(Asrc, H_, rowt + 48, kk, lane), bf, c3);
  }
  int n = lane & 15, mb = (lane >> 4) * 8;
  float bv = bias[colt + n];
  v8f acc[4] = {c0, c1, c2, c3};
#pragma unroll
  for (int t = 0; t < 4; ++t)
#pragma unroll
    for (int j = 0; j < 8; ++j)
      outp[((size_t)(rowt + t * 16 + mb + j)) * H_ + colt + n] =
          (__bf16)fmaxf(acc[t][j] + bv, 0.0f);
}

// h = relu(fused @ W_h1 + b_h1)  [BR,768]x[768,512] -> bf16. grid (HID/64, BR/64)
__global__ void h1_gemm(const __bf16* __restrict__ A, const __bf16* __restrict__ Bt,
                        const float* __restrict__ bias, __bf16* __restrict__ outp) {
  int wave = threadIdx.x >> 5, lane = threadIdx.x & 31;
  int colt = blockIdx.x * 64 + wave * 16;
  int rowt = blockIdx.y * 64;
  v8f c0 = {}, c1 = {}, c2 = {}, c3 = {};
  for (int k = 0; k < H_; k += 32) {
    v16bf bf = load_Bt_frag(Bt, H_, k, colt, lane);
    c0 = WMMA_BF16(load_A_frag(A, H_, rowt +  0, k, lane), bf, c0);
    c1 = WMMA_BF16(load_A_frag(A, H_, rowt + 16, k, lane), bf, c1);
    c2 = WMMA_BF16(load_A_frag(A, H_, rowt + 32, k, lane), bf, c2);
    c3 = WMMA_BF16(load_A_frag(A, H_, rowt + 48, k, lane), bf, c3);
  }
  int n = lane & 15, mb = (lane >> 4) * 8;
  float bv = bias[colt + n];
  v8f acc[4] = {c0, c1, c2, c3};
#pragma unroll
  for (int t = 0; t < 4; ++t)
#pragma unroll
    for (int j = 0; j < 8; ++j)
      outp[((size_t)(rowt + t * 16 + mb + j)) * HID_ + colt + n] =
          (__bf16)fmaxf(acc[t][j] + bv, 0.0f);
}

// logits = h @ W_h2 + b_h2  [BR,512]x[512,9] -> f32 [BR,9]
// Wh2t is [16,512] zero-padded; WMMA on full 16-wide tile (EXEC all-1s),
// store masks n<9. grid (BR/256), wave takes rows bx*256 + wave*64.
__global__ void h2_gemm(const __bf16* __restrict__ A, const __bf16* __restrict__ Bt,
                        const float* __restrict__ bias, float* __restrict__ outp) {
  int wave = threadIdx.x >> 5, lane = threadIdx.x & 31;
  int rowt = blockIdx.x * 256 + wave * 64;
  v8f c0 = {}, c1 = {}, c2 = {}, c3 = {};
  for (int k = 0; k < HID_; k += 32) {
    v16bf bf = load_Bt_frag(Bt, HID_, k, 0, lane);
    c0 = WMMA_BF16(load_A_frag(A, HID_, rowt +  0, k, lane), bf, c0);
    c1 = WMMA_BF16(load_A_frag(A, HID_, rowt + 16, k, lane), bf, c1);
    c2 = WMMA_BF16(load_A_frag(A, HID_, rowt + 32, k, lane), bf, c2);
    c3 = WMMA_BF16(load_A_frag(A, HID_, rowt + 48, k, lane), bf, c3);
  }
  int n = lane & 15, mb = (lane >> 4) * 8;
  v8f acc[4] = {c0, c1, c2, c3};
  if (n < NC_) {
    float bv = bias[n];
#pragma unroll
    for (int t = 0; t < 4; ++t)
#pragma unroll
      for (int j = 0; j < 8; ++j)
        outp[((size_t)(rowt + t * 16 + mb + j)) * NC_ + n] = acc[t][j] + bv;
  }
}

// ---------------------------------------------------------------------------
extern "C" void kernel_launch(void* const* d_in, const int* in_sizes, int n_in,
                              void* d_out, int out_size, void* d_ws, size_t ws_size,
                              hipStream_t stream) {
  (void)in_sizes; (void)n_in; (void)out_size; (void)ws_size;
  const float* seq   = (const float*)d_in[0];
  const float* visf  = (const float*)d_in[1];
  const float* roi   = (const float*)d_in[2];
  const int*   bbox  = (const int*)d_in[3];
  const int*   amask = (const int*)d_in[4];
  const float* Wproj = (const float*)d_in[5];
  const float* bproj = (const float*)d_in[6];
  const float* Wfuse = (const float*)d_in[7];
  const float* bfuse = (const float*)d_in[8];
  const float* Wh1   = (const float*)d_in[9];
  const float* bh1   = (const float*)d_in[10];
  const float* Wh2   = (const float*)d_in[11];
  const float* bh2   = (const float*)d_in[12];
  float* out = (float*)d_out;

  // Workspace carve (~171 MB)
  char* ws = (char*)d_ws;
  size_t off = 0;
  auto take = [&](size_t bytes) -> char* {
    char* p = ws + off;
    off = (off + bytes + 255) & ~(size_t)255;
    return p;
  };
  __bf16* seqt   = (__bf16*)take((size_t)B_ * H_ * S_ * 2);   // [B][H,S]
  __bf16* visb   = (__bf16*)take((size_t)B_ * R_ * VD_ * 2);  // [BR,VD] row-major A
  __bf16* wprojt = (__bf16*)take((size_t)H_ * VD_ * 2);       // [H,VD]
  __bf16* wfuset = (__bf16*)take((size_t)H_ * 2 * H_ * 2);    // [H,2H]
  __bf16* wh1t   = (__bf16*)take((size_t)HID_ * H_ * 2);      // [512,768]
  __bf16* wh2t   = (__bf16*)take((size_t)NCP_ * HID_ * 2);    // [16,512]
  __bf16* maskb  = (__bf16*)take((size_t)B_ * R_ * S_ * 2);
  float*  invc   = (float*)take((size_t)B_ * R_ * 4);
  __bf16* textb  = (__bf16*)take((size_t)B_ * R_ * H_ * 2);
  __bf16* vispb  = (__bf16*)take((size_t)B_ * R_ * H_ * 2);
  __bf16* fusedb = (__bf16*)take((size_t)B_ * R_ * H_ * 2);
  __bf16* hb     = (__bf16*)take((size_t)B_ * R_ * HID_ * 2);

  // 1) conversions: activations (row-major) + all B operands transposed to [N,K]
  long n = (long)B_ * R_ * VD_;
  cvt_f32_bf16<<<(int)((n + 511) / 512), 512, 0, stream>>>(visf, visb, n);
  // seq [S,H] -> seqt [H,S] per batch
  transpose_cvt<<<dim3(H_ / 32, S_ / 32, B_), 256, 0, stream>>>(
      seq, seqt, S_, H_, (long)S_ * H_, (long)H_ * S_);
  // W_proj [VD,H] -> [H,VD]
  transpose_cvt<<<dim3(H_ / 32, VD_ / 32, 1), 256, 0, stream>>>(
      Wproj, wprojt, VD_, H_, 0, 0);
  // W_fuse [1536,768] -> [768,1536]
  transpose_cvt<<<dim3(H_ / 32, (2 * H_) / 32, 1), 256, 0, stream>>>(
      Wfuse, wfuset, 2 * H_, H_, 0, 0);
  // W_h1 [768,512] -> [512,768]
  transpose_cvt<<<dim3(HID_ / 32, H_ / 32, 1), 256, 0, stream>>>(
      Wh1, wh1t, H_, HID_, 0, 0);
  // W_h2 [512,9] -> [16,512] zero-padded
  cvt_pad_wh2_t<<<(NCP_ * HID_ + 255) / 256, 256, 0, stream>>>(Wh2, wh2t);

  // 2) containment mask + inverse counts
  mask_kernel<<<B_ * R_, S_, 0, stream>>>(bbox, roi, amask, maskb, invc);

  // 3) masked mean-pool as batched WMMA GEMM
  pool_gemm<<<dim3(H_ / 64, R_ / 64, B_), 128, 0, stream>>>(maskb, seqt, invc, textb);

  // 4) visual projection GEMM
  vis_gemm<<<dim3(H_ / 64, (B_ * R_) / 64), 128, 0, stream>>>(visb, wprojt, bproj, vispb);

  // 5) fusion GEMM (K=1536 concat) + ReLU
  fuse_gemm<<<dim3(H_ / 64, (B_ * R_) / 64), 128, 0, stream>>>(textb, vispb, wfuset,
                                                               bfuse, fusedb);

  // 6) hidden GEMM + ReLU
  h1_gemm<<<dim3(HID_ / 64, (B_ * R_) / 64), 128, 0, stream>>>(fusedb, wh1t, bh1, hb);

  // 7) classifier GEMM -> f32 logits
  h2_gemm<<<(B_ * R_) / 256, 128, 0, stream>>>(hb, wh2t, bh2, out);
}
